// BatchedArcFaceLayer_23527830848003
// MI455X (gfx1250) — matrix operand order repaired and checked by reading, not compiled, not run on hardware
//
#include <hip/hip_runtime.h>
#include <hip/hip_bf16.h>

typedef __attribute__((ext_vector_type(16))) __bf16 v16bf;
typedef __attribute__((ext_vector_type(8)))  float  v8f;

#define IN_F   512
#define OUT_F  100000
#define NBATCH 1024
#define NPAD   100096      /* 391 * 256 */
#define SCALE_F 30.0f
#define COS_M  0.8775825618903728f
#define SIN_M  0.4794255386042030f
#define TH_C  (-0.8775825618903728f)
#define MM_C   0.2397127693021015f

// ---------------------------------------------------------------------------
// Kernel 1: normalize x rows -> bf16, written directly in WMMA A-fragment
// layout. A buffer: [64 m-tiles][16 k-tiles][512 bf16 per 16x32 fragment],
// fragment element order = lane*16 + inlane so the GEMM reads one contiguous
// v16bf (32B) per lane.
// ---------------------------------------------------------------------------
__global__ __launch_bounds__(256) void norm_x_kernel(
    const float* __restrict__ x, __bf16* __restrict__ A) {
  const int row  = blockIdx.x * 8 + (threadIdx.x >> 5);
  const int lane = threadIdx.x & 31;
  const float* px = x + (size_t)row * IN_F + lane * 16;

  float v[16];
  float s = 0.0f;
#pragma unroll
  for (int i = 0; i < 16; ++i) { v[i] = px[i]; s += v[i] * v[i]; }
#pragma unroll
  for (int m = 16; m >= 1; m >>= 1) s += __shfl_xor(s, m, 32);
  const float inv = 1.0f / fmaxf(sqrtf(s), 1e-12f);

  const int mt = row >> 4, mloc = row & 15;
#pragma unroll
  for (int i = 0; i < 16; ++i) {
    const int k  = lane * 16 + i;
    const int kt = k >> 5, kp = k & 31;
    const int dlane  = mloc + ((kp >> 4) << 4);
    const int inlane = kp & 15;
    A[((size_t)(mt * 16 + kt)) * 512 + dlane * 16 + inlane] = (__bf16)(v[i] * inv);
  }
}

// ---------------------------------------------------------------------------
// Kernel 2: normalize weight rows -> bf16 row-major [NPAD][512], pad rows = 0.
// ---------------------------------------------------------------------------
__global__ __launch_bounds__(256) void norm_w_kernel(
    const float* __restrict__ w, __bf16* __restrict__ Wn) {
  const int row  = blockIdx.x * 8 + (threadIdx.x >> 5);
  const int lane = threadIdx.x & 31;
  v16bf out;
  if (row < OUT_F) {
    const float* pw = w + (size_t)row * IN_F + lane * 16;
    float v[16];
    float s = 0.0f;
#pragma unroll
    for (int i = 0; i < 16; ++i) { v[i] = pw[i]; s += v[i] * v[i]; }
#pragma unroll
    for (int m = 16; m >= 1; m >>= 1) s += __shfl_xor(s, m, 32);
    const float inv = 1.0f / fmaxf(sqrtf(s), 1e-12f);
#pragma unroll
    for (int i = 0; i < 16; ++i) out[i] = (__bf16)(v[i] * inv);
  } else {
#pragma unroll
    for (int i = 0; i < 16; ++i) out[i] = (__bf16)0.0f;
  }
  *(v16bf*)(Wn + (size_t)row * IN_F + lane * 16) = out;
}

// ---------------------------------------------------------------------------
// Kernel 3: C[1024, 100000] = 30 * (A @ Wn^T) via v_wmma_f32_16x16x32_bf16.
// Block = 8 waves, all sharing the same M range (A slice = 64KB = WGP$).
// Block tile 64 x 256; wave tile 64(M) x 32(N): per K-step 4 A-frag loads,
// 2 B-frag loads, 8 WMMAs -> B (the streamed operand) amortized over 64 rows.
// ---------------------------------------------------------------------------
__global__ __launch_bounds__(256) void gemm_kernel(
    const __bf16* __restrict__ A,   // swizzled fragments
    const __bf16* __restrict__ Wn,  // [NPAD][512]
    float* __restrict__ C) {
  const int lane = threadIdx.x & 31;
  const int wave = threadIdx.x >> 5;       // 0..7 -> N partition
  const int M0 = blockIdx.y * 64;          // same for all waves in block
  const int N0 = blockIdx.x * 256 + wave * 32;
  const int mt0 = M0 >> 4;

  v8f acc[4][2] = {};

  const int nll = lane & 15;               // column within 16-tile
  const int khalf = (lane >> 4) << 4;      // lanes 16-31 hold K=16..31 of chunk

#pragma unroll 2
  for (int kt = 0; kt < IN_F / 32; ++kt) {
    v16bf a[4];
#pragma unroll
    for (int i = 0; i < 4; ++i)
      a[i] = *(const v16bf*)(A + ((size_t)((mt0 + i) * 16 + kt)) * 512 + lane * 16);
#pragma unroll
    for (int j = 0; j < 2; ++j) {
      const int n = N0 + j * 16 + nll;
      const __bf16* p = Wn + (size_t)n * IN_F + kt * 32 + khalf;
      const v16bf b = *(const v16bf*)p;
      __builtin_prefetch(p + 32, 0, 1);    // next K-slice -> global_prefetch
#pragma unroll
      for (int i = 0; i < 4; ++i)
        acc[i][j] = __builtin_amdgcn_wmma_f32_16x16x32_bf16(
            false, a[i], false, b, (short)0, acc[i][j], false, false);
    }
  }

  // C/D layout: VGPR r -> row M=r (lanes 0-15) / M=r+8 (lanes 16-31), N=lane&15
  const int mhalf = (lane >> 4) << 3;
#pragma unroll
  for (int i = 0; i < 4; ++i) {
#pragma unroll
    for (int j = 0; j < 2; ++j) {
      const int n = N0 + j * 16 + nll;
      if (n < OUT_F) {
        const int mbase = M0 + i * 16 + mhalf;
#pragma unroll
        for (int r = 0; r < 8; ++r) {
          C[(size_t)(mbase + r) * OUT_F + n] = acc[i][j][r] * SCALE_F;
        }
      }
    }
  }
}

// ---------------------------------------------------------------------------
// Kernel 4: fp32-exact margin fixup at out[row, label]. One wave per row.
// ---------------------------------------------------------------------------
__global__ __launch_bounds__(32) void fix_labels_kernel(
    const float* __restrict__ x, const float* __restrict__ w,
    const int* __restrict__ label, float* __restrict__ C) {
  const int row  = blockIdx.x;
  const int lane = threadIdx.x;
  const int lab  = label[row];

  const float* px = x + (size_t)row * IN_F + lane * 16;
  const float* pw = w + (size_t)lab * IN_F + lane * 16;
  float sx = 0.0f, sw = 0.0f, d = 0.0f;
#pragma unroll
  for (int i = 0; i < 16; ++i) {
    const float a = px[i], b = pw[i];
    sx += a * a; sw += b * b; d += a * b;
  }
#pragma unroll
  for (int m = 16; m >= 1; m >>= 1) {
    sx += __shfl_xor(sx, m, 32);
    sw += __shfl_xor(sw, m, 32);
    d  += __shfl_xor(d,  m, 32);
  }
  if (lane == 0) {
    const float c = d / (fmaxf(sqrtf(sx), 1e-12f) * fmaxf(sqrtf(sw), 1e-12f));
    const float s = sqrtf(fmaxf(1.0f - c * c, 0.0f));
    float phi = c * COS_M - s * SIN_M;
    phi = (c > TH_C) ? phi : (c - MM_C);
    C[(size_t)row * OUT_F + lab] = phi * SCALE_F;
  }
}

// ---------------------------------------------------------------------------
extern "C" void kernel_launch(void* const* d_in, const int* in_sizes, int n_in,
                              void* d_out, int out_size, void* d_ws, size_t ws_size,
                              hipStream_t stream) {
  const float* x   = (const float*)d_in[0];
  const float* w   = (const float*)d_in[1];
  const int*   lab = (const int*)d_in[2];
  float* C = (float*)d_out;

  // workspace: Wn bf16 [NPAD*512] then A fragments bf16 [1024*512]
  __bf16* Wn = (__bf16*)d_ws;
  __bf16* Ax = (__bf16*)((char*)d_ws + (size_t)NPAD * IN_F * sizeof(__bf16));

  norm_x_kernel<<<NBATCH / 8, 256, 0, stream>>>(x, Ax);
  norm_w_kernel<<<NPAD / 8, 256, 0, stream>>>(w, Wn);

  dim3 grid(NPAD / 256, NBATCH / 64);
  gemm_kernel<<<grid, 256, 0, stream>>>(Ax, Wn, C);

  fix_labels_kernel<<<NBATCH, 32, 0, stream>>>(x, w, lab, C);
}
